// ConvolutionalCapsules_66477503808119
// MI455X (gfx1250) — matrix operand order repaired and checked by reading, not compiled, not run on hardware
//
#include <hip/hip_runtime.h>
#include <stdint.h>

// Problem constants (from reference)
#define B_    8
#define IC_   16
#define ID_   16
#define OC_   8
#define OD_   16
#define HH    32
#define WW    32
#define CIN   64            // ID*4
#define M_TOT 512           // OC*OD*4  (rows m = o*4 + r)
#define N_TOT 1024          // H*W
#define K_TOT 576           // CIN*9
#define PADH  34            // 32 + 2*pad
#define KPAN  64            // K panel staged in LDS per double-buffer slot

typedef __attribute__((ext_vector_type(16))) __bf16 v16bf;
typedef __attribute__((ext_vector_type(8)))  float  v8f;

static __device__ __forceinline__ unsigned short f2bf(float f) {
  union { float f; uint32_t u; } v; v.f = f;
  uint32_t u = v.u;
  uint32_t lsb = (u >> 16) & 1u;
  u += 0x7FFFu + lsb;              // round-to-nearest-even
  return (unsigned short)(u >> 16);
}

// ------------------------------------------------------------------
// Kernel 1: xs[b,c=id*4+rot, y, x] = sum_ic x[b,ic,id,rot,y-1,x-1]
// stored zero-padded (34x34) as bf16. One block per (b,c).
// ------------------------------------------------------------------
__global__ void xs_reduce_kernel(const float* __restrict__ x,
                                 unsigned short* __restrict__ xs_pad) {
  const int bc = blockIdx.x;                 // 0..511
  const int b  = bc >> 6;
  const int c  = bc & 63;                    // c = id*4 + rot
  const int id = c >> 2, rot = c & 3;
  const float* xb = x + (((size_t)b * IC_ * ID_ + id) * 4 + rot) * (HH * WW);
  unsigned short* outp = xs_pad + (size_t)bc * (PADH * PADH);
  for (int idx = threadIdx.x; idx < PADH * PADH; idx += blockDim.x) {
    const int y = idx / PADH, xx = idx % PADH;
    float s = 0.f;
    if (y >= 1 && y <= HH && xx >= 1 && xx <= WW) {
      const int off = (y - 1) * WW + (xx - 1);
#pragma unroll
      for (int ic = 0; ic < IC_; ++ic)
        s += xb[(size_t)ic * (ID_ * 4 * HH * WW) + off];
    }
    outp[idx] = f2bf(s);
  }
}

// ------------------------------------------------------------------
// Kernel 2: P4-transformed filter, row-major [M_TOT][K_TOT] bf16.
// filt[(o*4+r)][(i*4+s)*9 + a*3 + b] = rot90^r( w[o, i, (s-r)%4] )[a,b]
// ------------------------------------------------------------------
__global__ void filt_build_kernel(const float* __restrict__ w,
                                  unsigned short* __restrict__ filt) {
  const int t = blockIdx.x * blockDim.x + threadIdx.x;
  if (t >= M_TOT * K_TOT) return;
  const int m = t / K_TOT, k = t % K_TOT;
  const int o = m >> 2, r = m & 3;
  const int c = k / 9, rem = k % 9, a = rem / 3, bq = rem % 3;
  const int i = c >> 2, s = c & 3;
  const int ssrc = (s - r) & 3;              // roll on rotation axis
  int ka, kb;                                // rot90 applied r times (CCW)
  switch (r) {
    case 0:  ka = a;      kb = bq;     break;
    case 1:  ka = bq;     kb = 2 - a;  break;
    case 2:  ka = 2 - a;  kb = 2 - bq; break;
    default: ka = 2 - bq; kb = a;      break;
  }
  const float v = w[(((size_t)o * ID_ + i) * 4 + ssrc) * 9 + ka * 3 + kb];
  filt[t] = f2bf(v);
}

// ------------------------------------------------------------------
// Kernel 3: im2col, per batch: Bcol[b][n=h*32+w][k=c*9+a*3+bb] (bf16)
// ------------------------------------------------------------------
__global__ void im2col_kernel(const unsigned short* __restrict__ xs_pad,
                              unsigned short* __restrict__ bcol) {
  const int t = blockIdx.x * blockDim.x + threadIdx.x;  // < 8*1024*576
  if (t >= B_ * N_TOT * K_TOT) return;
  const int b   = t / (N_TOT * K_TOT);
  const int rem = t - b * (N_TOT * K_TOT);
  const int n = rem / K_TOT, k = rem % K_TOT;
  const int h = n >> 5, wv = n & 31;
  const int c = k / 9, r9 = k % 9, a = r9 / 3, bb = r9 % 3;
  bcol[t] = xs_pad[(((size_t)b * CIN + c) * PADH + (h + a)) * PADH + (wv + bb)];
}

// ------------------------------------------------------------------
// Kernel 4: per-batch GEMM C = filt x Bcol via WMMA bf16 -> f32.
// Block tile 64M x 64N, 4 waves (one 16M row-band each) share a B panel
// staged in LDS via double-buffered global_load_async_to_lds_b128
// (ASYNCcnt pipeline). Epilogue: bias (16*conv_b[o]) + squash over the
// 4 rotation rows (which live in one lane's accumulator VGPRs).
// ------------------------------------------------------------------
__global__ void __launch_bounds__(128)
gemm_squash_kernel(const unsigned short* __restrict__ filt,
                   const unsigned short* __restrict__ bcol,
                   const float* __restrict__ bias,
                   float* __restrict__ out) {
  __shared__ unsigned short Bpan[2][KPAN * 64];   // [buf][n_local*64 + k_local], 16 KB

  const int lane = threadIdx.x & 31;
  const int wave = threadIdx.x >> 5;          // 0..3
  const int blk  = blockIdx.x;                // 0..1023 = 8 * 8 * 16
  const int b    = blk >> 7;                  // / (8*16)
  const int rem  = blk & 127;
  const int mblk = rem >> 4;                  // 0..7   (64 M rows per block)
  const int nblk = rem & 15;                  // 0..15  (64 N cols per block)
  const int m_base = mblk * 64 + wave * 16;
  const int n_base = nblk * 64;
  const int half = lane >> 4;                 // 0 or 1
  const int l15  = lane & 15;

  // A fragment: lane row M = l15; K chunks {half*8+0..7, 16+half*8+0..7}
  const unsigned short* Abase =
      filt + (size_t)(m_base + l15) * K_TOT + half * 8;
  // B panel source for this block: rows n_base..n_base+63, k contiguous
  const unsigned short* Bsrc =
      bcol + ((size_t)b * N_TOT + n_base) * K_TOT;

  // Async issue of one 64N x 64K panel into LDS buffer `buf`.
  // 64 rows x 128 B/row = 512 16-B lane-transfers = 4 per thread.
  const int chunk = (threadIdx.x & 7) * 8;    // k elem offset of 16-B chunk
  const int row0  = threadIdx.x >> 3;         // 0..15
  auto issue_panel = [&](int kb, int buf) {
    const unsigned short* g0 = Bsrc + kb * KPAN + chunk;
#pragma unroll
    for (int p = 0; p < 4; ++p) {
      const int row = row0 + p * 16;          // n_local 0..63
      const uint64_t ga = (uint64_t)(uintptr_t)(g0 + (size_t)row * K_TOT);
      const uint32_t la =
          (uint32_t)(uintptr_t)(&Bpan[buf][row * KPAN + chunk]);
      asm volatile("global_load_async_to_lds_b128 %0, %1, off"
                   :: "v"(la), "v"(ga) : "memory");
    }
  };

  v8f zero = {};
  v8f acc[4];
#pragma unroll
  for (int j = 0; j < 4; ++j) acc[j] = zero;

  union Frag { uint4 q[2]; v16bf v; };

  issue_panel(0, 0);
  for (int kb = 0; kb < K_TOT / KPAN; ++kb) { // 9 panels
    const int buf = kb & 1;
    if (kb + 1 < K_TOT / KPAN) {
      issue_panel(kb + 1, buf ^ 1);
      asm volatile("s_wait_asynccnt 0x4" ::: "memory");  // panel kb landed
    } else {
      asm volatile("s_wait_asynccnt 0x0" ::: "memory");
    }
    __syncthreads();                          // all waves' copies visible

    const unsigned short* pan = &Bpan[buf][0];
#pragma unroll
    for (int ks = 0; ks < KPAN; ks += 32) {
      Frag af;
      const uint4* ap = (const uint4*)(Abase + kb * KPAN + ks);
      af.q[0] = ap[0];                        // K = +half*8 + 0..7
      af.q[1] = ap[2];                        // K = +16 + half*8 + 0..7
      if (ks == 0 && kb + 1 < K_TOT / KPAN)
        __builtin_prefetch(Abase + (kb + 1) * KPAN, 0, 1);  // global_prefetch_b8
#pragma unroll
      for (int j = 0; j < 4; ++j) {
        Frag bf;                              // B: lane col N=l15, K={half*8+0..7, 16+half*8+0..7}
        const uint4* bp =
            (const uint4*)(pan + (j * 16 + l15) * KPAN + ks + half * 8);
        bf.q[0] = bp[0];
        bf.q[1] = bp[2];
        acc[j] = __builtin_amdgcn_wmma_f32_16x16x32_bf16(
            false, af.v, false, bf.v, (short)0, acc[j], false, false);
      }
    }
    __syncthreads();                          // done reading buf before reuse
  }

  // Bias per output channel o = m/4 (16x from the IC sum), then squash over
  // the 4 rotation rows. C layout: VGPR v <-> row v + 8*half (lane l15 = N).
  const int o0 = (m_base >> 2) + 2 * half;
  const float bg0 = 16.0f * bias[o0];
  const float bg1 = 16.0f * bias[o0 + 1];

#pragma unroll
  for (int j = 0; j < 4; ++j) {
    const int n = n_base + j * 16 + l15;
    float* op = out + ((size_t)b * M_TOT + m_base + half * 8) * N_TOT + n;
    {
      const float t0 = acc[j][0] + bg0, t1 = acc[j][1] + bg0;
      const float t2 = acc[j][2] + bg0, t3 = acc[j][3] + bg0;
      const float n2 = t0 * t0 + t1 * t1 + t2 * t2 + t3 * t3;
      const float nr = __builtin_sqrtf(n2);
      const float sc = n2 / ((1.0f + n2) * (nr + 1e-8f));
      op[0 * N_TOT] = sc * t0; op[1 * N_TOT] = sc * t1;
      op[2 * N_TOT] = sc * t2; op[3 * N_TOT] = sc * t3;
    }
    {
      const float t0 = acc[j][4] + bg1, t1 = acc[j][5] + bg1;
      const float t2 = acc[j][6] + bg1, t3 = acc[j][7] + bg1;
      const float n2 = t0 * t0 + t1 * t1 + t2 * t2 + t3 * t3;
      const float nr = __builtin_sqrtf(n2);
      const float sc = n2 / ((1.0f + n2) * (nr + 1e-8f));
      op[4 * N_TOT] = sc * t0; op[5 * N_TOT] = sc * t1;
      op[6 * N_TOT] = sc * t2; op[7 * N_TOT] = sc * t3;
    }
  }
}

// ------------------------------------------------------------------
extern "C" void kernel_launch(void* const* d_in, const int* in_sizes, int n_in,
                              void* d_out, int out_size, void* d_ws, size_t ws_size,
                              hipStream_t stream) {
  const float* x      = (const float*)d_in[0];  // (8,16,16,4,32,32)
  const float* conv_w = (const float*)d_in[1];  // (128,16,4,3,3)
  const float* conv_b = (const float*)d_in[2];  // (128,)
  // d_in[3]=ln_gamma, d_in[4]=ln_beta, d_in[5]=k, d_in[6]=ITER are unused:
  // softmax over a size-1 axis makes all routing weights exactly 1.
  float* out = (float*)d_out;                   // (8,8,16,4,32,32) fp32

  char* ws = (char*)d_ws;
  unsigned short* xs_pad = (unsigned short*)ws;                       // 8*64*34*34   = 591,872 bf16
  unsigned short* filt   = (unsigned short*)(ws + 1183744);           // 512*576      = 294,912 bf16
  unsigned short* bcol   = (unsigned short*)(ws + 1183744 + 589824);  // 8*1024*576   = 4,718,592 bf16

  xs_reduce_kernel<<<B_ * CIN, 256, 0, stream>>>(x, xs_pad);
  filt_build_kernel<<<(M_TOT * K_TOT + 255) / 256, 256, 0, stream>>>(conv_w, filt);
  im2col_kernel<<<(B_ * N_TOT * K_TOT + 255) / 256, 256, 0, stream>>>(xs_pad, bcol);
  gemm_squash_kernel<<<1024, 128, 0, stream>>>(filt, bcol, conv_b, out);
  (void)in_sizes; (void)n_in; (void)out_size; (void)ws_size;
}
